// SpectralBandExtractor_8014408974967
// MI455X (gfx1250) — matrix operand order repaired and checked by reading, not compiled, not run on hardware
//
#include <hip/hip_runtime.h>
#include <math.h>

typedef __attribute__((ext_vector_type(16))) __bf16 v16bf;
typedef __attribute__((ext_vector_type(8)))  float  v8f;

// ---------- helpers ----------
static __device__ inline __bf16 f2bf(float f) {
  unsigned u = __builtin_bit_cast(unsigned, f);
  unsigned r = (u + 0x7fffu + ((u >> 16) & 1u)) >> 16;   // RNE truncate to bf16
  return __builtin_bit_cast(__bf16, (unsigned short)r);
}

static __device__ inline v8f wmma_bf16(v16bf a, v16bf b, v8f c) {
  // (neg_a, A, neg_b, B, c_mod, C, reuse_a, reuse_b)
  return __builtin_amdgcn_wmma_f32_16x16x32_bf16(false, a, false, b, (short)0, c, false, false);
}

// Async global->LDS 16-byte copy (CDNA5 ASYNCcnt path; bypasses VGPRs).
// lds_off = workgroup-relative LDS byte offset (low 32 bits of generic ptr).
static __device__ inline void async_cp16(const __bf16* g, unsigned lds_off) {
  asm volatile("global_load_async_to_lds_b128 %0, %1, off"
               :: "v"(lds_off), "v"(g) : "memory");
}
static __device__ inline void wait_async() {
  asm volatile("s_wait_asynccnt 0x0" ::: "memory");
}
static __device__ inline unsigned lds_off(const void* p) {
  return (unsigned)(size_t)p;   // flat LDS address keeps byte offset in addr[31:0]
}

// A-matrix 16x32 bf16 fragment in contiguous k order (k = 16*half + s, one 32B run per lane)
static __device__ inline v16bf load_afrag(const __bf16* A, int lda, int lane) {
  int m = lane & 15, half = lane >> 4;
  v16bf a;
#pragma unroll
  for (int v = 0; v < 8; ++v) {
    int kb = half * 16 + 2 * v;
    a[2 * v]     = A[m * lda + kb];
    a[2 * v + 1] = A[m * lda + kb + 1];
  }
  return a;
}

// B-matrix 32x16 from a 1-D signal (implicit im2col): B[k][n] = sig[n + k]
static __device__ inline v16bf load_bfrag_sig(const __bf16* sig, int lane) {
  int n = lane & 15, kh = lane >> 4;
  v16bf b;
#pragma unroll
  for (int v = 0; v < 8; ++v) {
    int k = kh * 16 + 2 * v;
    b[2 * v]     = sig[n + k];
    b[2 * v + 1] = sig[n + k + 1];
  }
  return b;
}

// B-matrix 32x16 from a row-major [K x N] LDS matrix
static __device__ inline v16bf load_bfrag_mat(const __bf16* B, int ldb, int lane) {
  int n = lane & 15, kh = lane >> 4;
  v16bf b;
#pragma unroll
  for (int v = 0; v < 8; ++v) {
    int k = kh * 16 + 2 * v;
    b[2 * v]     = B[k * ldb + n];
    b[2 * v + 1] = B[(k + 1) * ldb + n];
  }
  return b;
}

// ---------- 0) restage weights to bf16 so all LDS tile fills are pure copies ----------
__global__ __launch_bounds__(256) void k_prep(const float* __restrict__ fw1,
                                              const float* __restrict__ fw2,
                                              const float* __restrict__ aw2,
                                              __bf16* __restrict__ fw1bf,
                                              __bf16* __restrict__ fw2bf,
                                              __bf16* __restrict__ aw2p) {
  int j = blockIdx.x * 256 + threadIdx.x;            // 8192 + 65536 + 65536
  if (j < 8192) {
    fw1bf[j] = f2bf(fw1[j]);
  } else if (j < 8192 + 65536) {
    int q = j - 8192;
    fw2bf[q] = f2bf(fw2[q]);
  } else {
    int q = j - 8192 - 65536;                        // aw2p[i][o][k] = aw2[o][i][k]
    int i = q >> 11, rem = q & 2047, o = rem >> 5, k = rem & 31;
    aw2p[q] = f2bf(aw2[(o * 32 + i) * 32 + k]);
  }
}

// ---------- 1) channel mean * filter_strength ----------
__global__ __launch_bounds__(256) void k_ca(const float* __restrict__ eeg,
                                            const float* __restrict__ fsp,
                                            float* __restrict__ ca) {
  int id = blockIdx.x * 256 + threadIdx.x;           // 16*1024
  int b = id >> 10, t = id & 1023;
  float s = 0.f;
#pragma unroll 8
  for (int c = 0; c < 64; ++c) s += eeg[((size_t)(b * 64 + c)) * 1024 + t];
  ca[id] = s * fsp[0] * (1.0f / 64.0f);
}

// ---------- 2) conv1 -> y1pad in bf16 (stride 1120, zero margins) ----------
__global__ __launch_bounds__(256) void k_conv1(const float* __restrict__ ca,
                                               const float* __restrict__ aw1,
                                               const float* __restrict__ ab1,
                                               __bf16* __restrict__ y1pad) {
  int id = blockIdx.x * 256 + threadIdx.x;           // 512 * 1120
  int row = id / 1120, tpad = id - row * 1120;
  int b = row >> 5, i = row & 31;
  int tp = tpad - 16;
  float v = 0.f;
  if (tp >= 0 && tp < 1025) {
    float s = ab1[i];
#pragma unroll 8
    for (int k = 0; k < 64; ++k) {
      int g = tp - 32 + k;
      if (g >= 0 && g < 1024) s += aw1[i * 64 + k] * ca[b * 1024 + g];
    }
    v = fmaxf(s, 0.f);
  }
  y1pad[id] = f2bf(v);
}

// ---------- 3) conv2 as WMMA GEMM, tiles filled by async DMA ----------
__global__ __launch_bounds__(128) void k_conv2(const __bf16* __restrict__ aw2p,
                                               const float* __restrict__ ab2,
                                               const __bf16* __restrict__ y1pad,
                                               float* __restrict__ h2) {
  __shared__ alignas(16) __bf16 sA[64 * 32];
  __shared__ alignas(16) __bf16 sSig[96];
  int tid = threadIdx.x, lane = tid & 31, w = tid >> 5;  // 4 waves = 4 M-tiles
  int nchunk = blockIdx.x, b = blockIdx.y;
  int t0 = nchunk * 64;
  unsigned sA_o = lds_off(sA), sS_o = lds_off(sSig);

  v8f acc[4];
#pragma unroll
  for (int nt = 0; nt < 4; ++nt)
#pragma unroll
    for (int r = 0; r < 8; ++r) acc[nt][r] = 0.f;

  for (int i = 0; i < 32; ++i) {                     // K-step = input channel
    __syncthreads();
    // weight slice aw2p[i][64][32]: 4KB = 256 x b128
#pragma unroll
    for (int c = 0; c < 2; ++c) {
      int ch = tid + c * 128;
      async_cp16(aw2p + i * 2048 + ch * 8, sA_o + ch * 16);
    }
    // signal slice y1pad[b][i][t0 .. t0+96): 192B = 12 x b128 (stride 1120 keeps it in-bounds)
    if (tid < 12)
      async_cp16(y1pad + ((size_t)(b * 32 + i)) * 1120 + t0 + tid * 8, sS_o + tid * 16);
    wait_async();
    __syncthreads();

    v16bf afr = load_afrag(&sA[(w * 16) * 32], 32, lane);
#pragma unroll
    for (int nt = 0; nt < 4; ++nt) {
      v16bf bfr = load_bfrag_sig(&sSig[nt * 16], lane);
      acc[nt] = wmma_bf16(afr, bfr, acc[nt]);
    }
  }
  int n = lane & 15, half = lane >> 4;
#pragma unroll
  for (int nt = 0; nt < 4; ++nt)
#pragma unroll
    for (int r = 0; r < 8; ++r) {
      int o = w * 16 + r + 8 * half;
      int t = t0 + nt * 16 + n;
      if (t < 1026)
        h2[((size_t)(b * 64 + o)) * 1026 + t] = fmaxf(acc[nt][r] + ab2[o], 0.f);
    }
}

// ---------- 4) adaptive pool 1026->1024 + 1x1 conv + sigmoid -> attn ----------
__global__ __launch_bounds__(256) void k_poolattn(const float* __restrict__ h2,
                                                  const float* __restrict__ aw3,
                                                  const float* __restrict__ ab3,
                                                  float* __restrict__ attn) {
  int id = blockIdx.x * 256 + threadIdx.x;           // 16*1024
  int b = id >> 10, t = id & 1023;
  int s = t + (t >= 512 ? 1 : 0);                    // exact torch bin mapping, count==2
  float acc = ab3[0];
#pragma unroll 8
  for (int o = 0; o < 64; ++o) {
    const float* p = &h2[((size_t)(b * 64 + o)) * 1026 + s];
    acc += aw3[o] * (p[0] + p[1]) * 0.5f;
  }
  attn[id] = 1.0f / (1.0f + expf(-acc));
}

// ---------- 5) fw1 conv + bn + relu + pairpool + channel-mean (fused, WMMA) ----------
__global__ __launch_bounds__(128) void k_fproj(const float* __restrict__ eeg,
                                               const float* __restrict__ fsp,
                                               const __bf16* __restrict__ fw1bf,
                                               const float* __restrict__ fb1,
                                               const float* __restrict__ bng,
                                               const float* __restrict__ bnb,
                                               const float* __restrict__ bnm,
                                               const float* __restrict__ bnv,
                                               const float* __restrict__ attn,
                                               __bf16* __restrict__ hbar) {
  __shared__ alignas(16) __bf16 sW[64 * 128];
  __shared__ float sScale[64], sShift[64];
  __shared__ alignas(16) __bf16 sX[208];
  int tid = threadIdx.x, lane = tid & 31, w = tid >> 5;  // 4 waves = 4 M-tiles
  int nchunk = blockIdx.x, b = blockIdx.y;
  int t0 = nchunk * 64;                                  // pooled-output chunk base
  float fs = fsp[0];

  // resident fw1 tile: 16KB = 1024 x b128 via async DMA
  unsigned sW_o = lds_off(sW);
#pragma unroll
  for (int c = 0; c < 8; ++c) {
    int ch = tid + c * 128;
    async_cp16(fw1bf + ch * 8, sW_o + ch * 16);
  }
  if (tid < 64) {
    float sc = bng[tid] * rsqrtf(bnv[tid] + 1e-5f);
    sScale[tid] = sc;
    sShift[tid] = (fb1[tid] - bnm[tid]) * sc + bnb[tid]; // folds conv bias into bn
  }
  wait_async();
  __syncthreads();

  v16bf af[4];                                           // fw1 A-fragments resident all c
#pragma unroll
  for (int ks = 0; ks < 4; ++ks) af[ks] = load_afrag(&sW[(w * 16) * 128 + ks * 32], 128, lane);

  int n = lane & 15, half = lane >> 4;
  float scl[8], shf[8];
#pragma unroll
  for (int r = 0; r < 8; ++r) {
    int o = w * 16 + r + 8 * half;
    scl[r] = sScale[o];
    shf[r] = sShift[o];
  }

  v8f P[4];
#pragma unroll
  for (int nt = 0; nt < 4; ++nt)
#pragma unroll
    for (int r = 0; r < 8; ++r) P[nt][r] = 0.f;

  for (int c = 0; c < 64; ++c) {
    __syncthreads();
    for (int j = tid; j < 208; j += 128) {               // x = eeg*fs*attn, zero-padded
      int g = t0 + j - 64;
      float v = 0.f;
      if (g >= 0 && g < 1024)
        v = eeg[((size_t)(b * 64 + c)) * 1024 + g] * fs * attn[b * 1024 + g];
      sX[j] = f2bf(v);
    }
    __syncthreads();

    float hb[5][8];                                      // 5 N-tiles: 80 conv cols, need 65
#pragma unroll
    for (int nt = 0; nt < 5; ++nt) {
      v8f acc;
#pragma unroll
      for (int r = 0; r < 8; ++r) acc[r] = 0.f;
#pragma unroll
      for (int ks = 0; ks < 4; ++ks) {
        v16bf bfr = load_bfrag_sig(&sX[nt * 16 + ks * 32], lane);
        acc = wmma_bf16(af[ks], bfr, acc);
      }
#pragma unroll
      for (int r = 0; r < 8; ++r)
        hb[nt][r] = fmaxf(acc[r] * scl[r] + shf[r], 0.f);  // bn + relu in C layout
    }
    // pairpool (t,t+1)/2 via cross-lane shuffle; edge column from next N-tile
#pragma unroll
    for (int nt = 0; nt < 4; ++nt)
#pragma unroll
      for (int r = 0; r < 8; ++r) {
        float cur = hb[nt][r];
        float s1 = __shfl(hb[nt][r], (n == 15) ? lane : lane + 1, 32);
        float s2 = __shfl(hb[nt + 1][r], half * 16, 32);
        float nb = (n == 15) ? s2 : s1;
        P[nt][r] += (cur + nb) * (0.5f / 64.0f);          // pool avg * channel mean
      }
  }
#pragma unroll
  for (int nt = 0; nt < 4; ++nt)
#pragma unroll
    for (int r = 0; r < 8; ++r) {
      int o = w * 16 + r + 8 * half;
      int t = t0 + nt * 16 + n;
      hbar[((size_t)(b * 64 + o)) * 1024 + t] = f2bf(P[nt][r]);
    }
}

// ---------- 6) feats[b] = fw2[1024x64] x hbar[b][64x1024] + fb2 (WMMA, async tiles) ----------
__global__ __launch_bounds__(128) void k_gemm(const __bf16* __restrict__ fw2bf,
                                              const float* __restrict__ fb2,
                                              const __bf16* __restrict__ hbar,
                                              float* __restrict__ feats) {
  __shared__ alignas(16) __bf16 sA[64 * 64];
  __shared__ alignas(16) __bf16 sB[64 * 64];
  int tid = threadIdx.x, lane = tid & 31, w = tid >> 5;
  int nchunk = blockIdx.x, mchunk = blockIdx.y, b = blockIdx.z;
  int m0 = mchunk * 64, n0 = nchunk * 64;
  unsigned sA_o = lds_off(sA), sB_o = lds_off(sB);

  // A tile: contiguous 8KB of fw2bf; B tile: 64 rows x 128B of hbar. 512 b128 each.
#pragma unroll
  for (int c = 0; c < 4; ++c) {
    int ch = tid + c * 128;
    async_cp16(fw2bf + (size_t)m0 * 64 + ch * 8, sA_o + ch * 16);
    int row = ch >> 3, c8 = ch & 7;
    async_cp16(hbar + ((size_t)(b * 64 + row)) * 1024 + n0 + c8 * 8, sB_o + ch * 16);
  }
  wait_async();
  __syncthreads();

  v16bf af0 = load_afrag(&sA[(w * 16) * 64 + 0], 64, lane);
  v16bf af1 = load_afrag(&sA[(w * 16) * 64 + 32], 64, lane);
  int n = lane & 15, half = lane >> 4;
#pragma unroll
  for (int nt = 0; nt < 4; ++nt) {
    v8f acc;
#pragma unroll
    for (int r = 0; r < 8; ++r) acc[r] = 0.f;
    v16bf b0 = load_bfrag_mat(&sB[0 * 64 + nt * 16], 64, lane);
    v16bf b1 = load_bfrag_mat(&sB[32 * 64 + nt * 16], 64, lane);
    acc = wmma_bf16(af0, b0, acc);
    acc = wmma_bf16(af1, b1, acc);
#pragma unroll
    for (int r = 0; r < 8; ++r) {
      int m = m0 + w * 16 + r + 8 * half;
      int t = n0 + nt * 16 + n;
      feats[((size_t)b * 1024 + m) * 1024 + t] = acc[r] + fb2[m];
    }
  }
}

// ---------- launch ----------
extern "C" void kernel_launch(void* const* d_in, const int* in_sizes, int n_in,
                              void* d_out, int out_size, void* d_ws, size_t ws_size,
                              hipStream_t stream) {
  const float* eeg = (const float*)d_in[0];
  const float* fs  = (const float*)d_in[1];
  const float* aw1 = (const float*)d_in[2];
  const float* ab1 = (const float*)d_in[3];
  const float* aw2 = (const float*)d_in[4];
  const float* ab2 = (const float*)d_in[5];
  const float* aw3 = (const float*)d_in[6];
  const float* ab3 = (const float*)d_in[7];
  const float* fw1 = (const float*)d_in[8];
  const float* fb1 = (const float*)d_in[9];
  const float* bng = (const float*)d_in[10];
  const float* bnb = (const float*)d_in[11];
  const float* bnm = (const float*)d_in[12];
  const float* bnv = (const float*)d_in[13];
  const float* fw2 = (const float*)d_in[14];
  const float* fb2 = (const float*)d_in[15];

  float* out  = (float*)d_out;
  float* attn = out + (size_t)16 * 1024 * 1024;   // second return value lives in d_out tail

  // workspace: fp32 region then bf16 region (16B aligned at the split)
  float* ws = (float*)d_ws;
  float*  ca    = ws;                              // 16*1024 f32
  float*  h2    = ca + 16 * 1024;                  // 16*64*1026 f32
  __bf16* bfb   = (__bf16*)(h2 + (size_t)16 * 64 * 1026);
  __bf16* y1pad = bfb;                             // 512*1120 bf16 (zero-padded)
  __bf16* hbar  = y1pad + 512 * 1120;              // 16*64*1024 bf16
  __bf16* fw1bf = hbar + (size_t)16 * 64 * 1024;   // 8192
  __bf16* fw2bf = fw1bf + 8192;                    // 65536
  __bf16* aw2p  = fw2bf + 65536;                   // 65536, [i][o][k]

  k_prep    <<<544, 256, 0, stream>>>(fw1, fw2, aw2, fw1bf, fw2bf, aw2p);
  k_ca      <<<64, 256, 0, stream>>>(eeg, fs, ca);
  k_conv1   <<<2240, 256, 0, stream>>>(ca, aw1, ab1, y1pad);
  k_conv2   <<<dim3(17, 16), 128, 0, stream>>>(aw2p, ab2, y1pad, h2);
  k_poolattn<<<64, 256, 0, stream>>>(h2, aw3, ab3, attn);
  k_fproj   <<<dim3(16, 16), 128, 0, stream>>>(eeg, fs, fw1bf, fb1, bng, bnb, bnm, bnv, attn, hbar);
  k_gemm    <<<dim3(16, 16, 16), 128, 0, stream>>>(fw2bf, fb2, hbar, out);
  (void)in_sizes; (void)n_in; (void)out_size; (void)ws_size;
}
